// PointnetFPModule_30468497998039
// MI455X (gfx1250) — matrix operand order repaired and checked by reading, not compiled, not run on hardware
//
#include <hip/hip_runtime.h>
#include <hip/hip_bf16.h>

#define BB 8
#define NN 8192
#define MM 2048
#define C1c 128
#define C2c 256
#define HHc 256
#define K1c 384   /* C1+C2 */

#define NT 128
#define PITCH_F 392   /* K1c + 8  (bf16 elements, keeps b128 alignment, avoids bank conflicts) */
#define PITCH_H 264   /* HHc + 8 */

typedef __attribute__((ext_vector_type(16))) __bf16 v16bf;
typedef __attribute__((ext_vector_type(8)))  float  v8f;

__device__ __forceinline__ unsigned short f2bf(float f) {
  union { float f; unsigned int u; } c; c.f = f;
  unsigned int u = c.u;
  unsigned int r = (u + 0x7FFFu + ((u >> 16) & 1u)) >> 16;  // round-to-nearest-even
  return (unsigned short)r;
}

// ---------------- kernel 0: convert W1/W2 to bf16 in workspace ----------------
__global__ void cvt_weights(const float* __restrict__ W1, const float* __restrict__ W2,
                            unsigned short* __restrict__ W1b, unsigned short* __restrict__ W2b) {
  int e = blockIdx.x * blockDim.x + threadIdx.x;
  if (e < HHc * K1c) W1b[e] = f2bf(W1[e]);
  if (e < HHc * HHc) W2b[e] = f2bf(W2[e]);
}

// ---------------- kernel 1: brute-force 3-NN + inverse-distance weights ----------------
__global__ void three_nn_kernel(const float* __restrict__ unknown,
                                const float* __restrict__ known,
                                int* __restrict__ idx_out,
                                float* __restrict__ w_out) {
  __shared__ float kxyz[MM * 3];                 // 24 KB
  const int b  = blockIdx.x >> 5;                // 32 blocks of 256 pts per batch
  const int n0 = (blockIdx.x & 31) * 256;
  const float* kb = known + (size_t)b * MM * 3;
  for (int e = threadIdx.x; e < MM * 3; e += blockDim.x) kxyz[e] = kb[e];
  __syncthreads();

  const int n = n0 + threadIdx.x;
  const float* up = unknown + ((size_t)b * NN + n) * 3;
  const float ux = up[0], uy = up[1], uz = up[2];

  float d0 = 1e30f, d1 = 1e30f, d2 = 1e30f;
  int j0 = 0, j1 = 0, j2 = 0;
  for (int i = 0; i < MM; ++i) {
    // uniform LDS reads -> broadcast, conflict-free
    float dx = ux - kxyz[3 * i + 0];
    float dy = uy - kxyz[3 * i + 1];
    float dz = uz - kxyz[3 * i + 2];
    float d = dx * dx + dy * dy + dz * dz;
    if (d < d2) {                                 // rarely taken after warm-up
      if (d < d0)      { d2 = d1; j2 = j1; d1 = d0; j1 = j0; d0 = d; j0 = i; }
      else if (d < d1) { d2 = d1; j2 = j1; d1 = d;  j1 = i; }
      else             { d2 = d;  j2 = i; }
    }
  }
  float r0 = 1.0f / (sqrtf(d0) + 1e-8f);
  float r1 = 1.0f / (sqrtf(d1) + 1e-8f);
  float r2 = 1.0f / (sqrtf(d2) + 1e-8f);
  float s = 1.0f / (r0 + r1 + r2);
  size_t base = ((size_t)b * NN + n) * 3;
  idx_out[base + 0] = j0; idx_out[base + 1] = j1; idx_out[base + 2] = j2;
  w_out[base + 0] = r0 * s; w_out[base + 1] = r1 * s; w_out[base + 2] = r2 * s;
}

// ---------------- kernel 2: fused interpolate + concat + MLP1 + MLP2 (bf16 WMMA) ----------------
__global__ void fp_mlp_kernel(const float* __restrict__ UF,          // (B,C1,N)
                              const float* __restrict__ KF,          // (B,C2,M)
                              const unsigned short* __restrict__ W1b,// (H,K1) bf16
                              const float* __restrict__ B1,
                              const unsigned short* __restrict__ W2b,// (H,H) bf16
                              const float* __restrict__ B2,
                              const int* __restrict__ idx,
                              const float* __restrict__ wgt,
                              float* __restrict__ out) {             // (B,H,N)
  extern __shared__ char smem[];
  unsigned short* featS = (unsigned short*)smem;            // [NT][PITCH_F]  col-major-per-column (K contiguous)
  unsigned short* hS    = featS + NT * PITCH_F;             // [NT][PITCH_H]
  int*   idxS = (int*)(hS + NT * PITCH_H);                  // [NT*3]
  float* wS   = (float*)(idxS + NT * 3);                    // [NT*3]

  const int b  = blockIdx.x >> 6;                 // 64 tiles of 128 cols per batch
  const int n0 = (blockIdx.x & 63) * NT;
  const int tid  = threadIdx.x;
  const int lane = tid & 31, wv = tid >> 5;
  const int half = lane >> 4, ln = lane & 15;

  // ---- stage 0: stage idx/weights for this column tile ----
  if (tid < NT * 3) {
    size_t base = ((size_t)b * NN + n0) * 3;
    idxS[tid] = idx[base + tid];
    wS[tid]   = wgt[base + tid];
  }
  __syncthreads();

  // ---- stage 1a: interpolated known_feats -> rows [0,C2) ----
  const float* KFb = KF + (size_t)b * C2c * MM;
  for (int e = tid; e < C2c * NT; e += blockDim.x) {
    int k = e >> 7, j = e & (NT - 1);
    const float* row = KFb + (size_t)k * MM;
    float v = wS[3 * j + 0] * row[idxS[3 * j + 0]]
            + wS[3 * j + 1] * row[idxS[3 * j + 1]]
            + wS[3 * j + 2] * row[idxS[3 * j + 2]];
    featS[j * PITCH_F + k] = f2bf(v);
  }
  // ---- stage 1b: unknow_feats -> rows [C2, C2+C1) ----
  const float* UFb = UF + (size_t)b * C1c * NN;
  for (int e = tid; e < C1c * NT; e += blockDim.x) {
    int c = e >> 7, j = e & (NT - 1);
    featS[j * PITCH_F + C2c + c] = f2bf(UFb[(size_t)c * NN + n0 + j]);
  }
  __syncthreads();

  // 16 waves: 8 row-groups (32 rows) x 2 col-groups (64 cols)
  const int rowBase = (wv & 7) * 32;
  const int colBase = (wv >> 3) * 64;

  union F16  { v16bf v; uint4 q[2]; };
  union Pack8 { uint4 q; unsigned short u[8]; };

  // ================= GEMM1: h = relu(W1 @ feat + b1) =================
  v8f acc[2][4];
#pragma unroll
  for (int rt = 0; rt < 2; ++rt)
#pragma unroll
    for (int v = 0; v < 8; ++v) {
      float bv = B1[rowBase + rt * 16 + half * 8 + v];
#pragma unroll
      for (int ct = 0; ct < 4; ++ct) acc[rt][ct][v] = bv;
    }

  for (int kk = 0; kk < K1c; kk += 32) {
    F16 a[2], bf[4];
#pragma unroll
    for (int rt = 0; rt < 2; ++rt) {
      // A layout (16-bit, 16x32): lane half 0 -> K 0..7 & 16..23, half 1 -> K 8..15 & 24..31
      const unsigned short* p = W1b + (size_t)(rowBase + rt * 16 + ln) * K1c + kk + half * 8;
      a[rt].q[0] = *(const uint4*)(p);
      a[rt].q[1] = *(const uint4*)(p + 16);
      __builtin_prefetch(p + 32, 0, 0);           // next K-chunk (L2-resident weights)
    }
#pragma unroll
    for (int ct = 0; ct < 4; ++ct) {
      // B layout (16-bit, 32x16): lane half 0 -> K 0..7 & 16..23, half 1 -> K 8..15 & 24..31
      const unsigned short* p = featS + (colBase + ct * 16 + ln) * PITCH_F + kk + half * 8;
      bf[ct].q[0] = *(const uint4*)(p);
      bf[ct].q[1] = *(const uint4*)(p + 16);
    }
#pragma unroll
    for (int rt = 0; rt < 2; ++rt)
#pragma unroll
      for (int ct = 0; ct < 4; ++ct)
        acc[rt][ct] = __builtin_amdgcn_wmma_f32_16x16x32_bf16(
            false, a[rt].v, false, bf[ct].v, (short)0, acc[rt][ct], false, false);
  }

  // relu -> bf16, store transposed into hS ([col][row], K contiguous for GEMM2's B frags)
#pragma unroll
  for (int rt = 0; rt < 2; ++rt)
#pragma unroll
    for (int ct = 0; ct < 4; ++ct) {
      Pack8 pk;
#pragma unroll
      for (int v = 0; v < 8; ++v) pk.u[v] = f2bf(fmaxf(acc[rt][ct][v], 0.0f));
      int col = colBase + ct * 16 + ln;
      int row = rowBase + rt * 16 + half * 8;     // D layout: VGPR v -> row v + 8*half
      *(uint4*)(hS + col * PITCH_H + row) = pk.q; // one ds_store_b128
    }
  __syncthreads();

  // ================= GEMM2: out = relu(W2 @ h + b2) =================
  v8f acc2[2][4];
#pragma unroll
  for (int rt = 0; rt < 2; ++rt)
#pragma unroll
    for (int v = 0; v < 8; ++v) {
      float bv = B2[rowBase + rt * 16 + half * 8 + v];
#pragma unroll
      for (int ct = 0; ct < 4; ++ct) acc2[rt][ct][v] = bv;
    }

  for (int kk = 0; kk < HHc; kk += 32) {
    F16 a[2], bf[4];
#pragma unroll
    for (int rt = 0; rt < 2; ++rt) {
      const unsigned short* p = W2b + (size_t)(rowBase + rt * 16 + ln) * HHc + kk + half * 8;
      a[rt].q[0] = *(const uint4*)(p);
      a[rt].q[1] = *(const uint4*)(p + 16);
    }
#pragma unroll
    for (int ct = 0; ct < 4; ++ct) {
      const unsigned short* p = hS + (colBase + ct * 16 + ln) * PITCH_H + kk + half * 8;
      bf[ct].q[0] = *(const uint4*)(p);
      bf[ct].q[1] = *(const uint4*)(p + 16);
    }
#pragma unroll
    for (int rt = 0; rt < 2; ++rt)
#pragma unroll
      for (int ct = 0; ct < 4; ++ct)
        acc2[rt][ct] = __builtin_amdgcn_wmma_f32_16x16x32_bf16(
            false, a[rt].v, false, bf[ct].v, (short)0, acc2[rt][ct], false, false);
  }

  // relu -> global out
  float* ob = out + (size_t)b * HHc * NN;
#pragma unroll
  for (int rt = 0; rt < 2; ++rt)
#pragma unroll
    for (int ct = 0; ct < 4; ++ct) {
      int colg = n0 + colBase + ct * 16 + ln;
#pragma unroll
      for (int v = 0; v < 8; ++v) {
        int row = rowBase + rt * 16 + half * 8 + v;
        ob[(size_t)row * NN + colg] = fmaxf(acc2[rt][ct][v], 0.0f);
      }
    }
}

extern "C" void kernel_launch(void* const* d_in, const int* in_sizes, int n_in,
                              void* d_out, int out_size, void* d_ws, size_t ws_size,
                              hipStream_t stream) {
  (void)in_sizes; (void)n_in; (void)out_size; (void)ws_size;
  const float* unknown      = (const float*)d_in[0];   // (B,N,3)
  const float* known        = (const float*)d_in[1];   // (B,M,3)
  const float* unknow_feats = (const float*)d_in[2];   // (B,C1,N)
  const float* known_feats  = (const float*)d_in[3];   // (B,C2,M)
  const float* W1           = (const float*)d_in[4];   // (H,K1)
  const float* b1           = (const float*)d_in[5];   // (H)
  const float* W2           = (const float*)d_in[6];   // (H,H)
  const float* b2           = (const float*)d_in[7];   // (H)
  float* out = (float*)d_out;

  // workspace layout (all offsets 16B aligned)
  char* ws = (char*)d_ws;
  int*            ws_idx = (int*)(ws);                          //  786432 B
  float*          ws_w   = (float*)(ws + 786432);               //  786432 B
  unsigned short* W1b    = (unsigned short*)(ws + 1572864);     //  196608 B
  unsigned short* W2b    = (unsigned short*)(ws + 1769472);     //  131072 B

  // 0) weight conversion to bf16
  cvt_weights<<<(HHc * K1c + 255) / 256, 256, 0, stream>>>(W1, W2, W1b, W2b);

  // 1) 3-NN + weights
  three_nn_kernel<<<BB * (NN / 256), 256, 0, stream>>>(unknown, known, ws_idx, ws_w);

  // 2) fused interpolate + concat + 2x GEMM (bf16 WMMA)
  const int smem_bytes = NT * PITCH_F * 2 + NT * PITCH_H * 2 + NT * 3 * 4 + NT * 3 * 4; // 171008
  static bool attr_set = false;
  (void)attr_set;
  hipFuncSetAttribute((const void*)fp_mlp_kernel,
                      hipFuncAttributeMaxDynamicSharedMemorySize, smem_bytes);
  fp_mlp_kernel<<<BB * (NN / NT), 512, smem_bytes, stream>>>(
      unknow_feats, known_feats, W1b, b1, W2b, b2, ws_idx, ws_w, out);
}
